// TemporalGNNCell_13297218748845
// MI455X (gfx1250) — matrix operand order repaired and checked by reading, not compiled, not run on hardware
//
#include <hip/hip_runtime.h>
#include <stdint.h>

// ---------------- problem constants (from reference) ----------------
constexpr int   BB    = 4;
constexpr int   NN    = 10000;
constexpr int   F_IN  = 128;
constexpr int   EE    = 160000;
constexpr int   HH    = 4;
constexpr int   CC    = 64;
constexpr int   HID   = 256;          // H*C
constexpr int   EP    = EE + NN;      // edges + self loops = 170000
constexpr int   ROWS  = BB * NN;      // 40000
constexpr long long TOT = (long long)ROWS * HID;   // 10,240,000 floats per output tensor
constexpr float NEG_SLOPE = 0.2f;
constexpr float LN_EPS    = 1e-5f;

// ---------------- workspace layout (bytes, all 256-aligned) ----------------
constexpr size_t OFF_XBF   = 0;                                   // x bf16 [40000][128]
constexpr size_t SZ_XBF    = (size_t)ROWS * F_IN * 2;             // 10,240,000
constexpr size_t OFF_WLIN  = OFF_XBF + SZ_XBF;                    // lin_w bf16 [256][128]
constexpr size_t SZ_WLIN   = (size_t)HID * F_IN * 2;              // 65,536
constexpr size_t OFF_WCAT  = OFF_WLIN + SZ_WLIN;                  // [W_ih|W_hh] bf16 [1024][512]
constexpr size_t SZ_WCAT   = (size_t)4 * HID * 2 * HID * 2;       // 1,048,576
constexpr size_t OFF_XT    = OFF_WCAT + SZ_WCAT;                  // xt f32 [40000][256]  (A2 bf16 aliases here later)
constexpr size_t SZ_XT     = (size_t)ROWS * HID * 4;              // 40,960,000
constexpr size_t OFF_AS    = OFF_XT + SZ_XT;                      // a_src f32 [40000][4]
constexpr size_t SZ_AH     = (size_t)ROWS * HH * 4;               // 640,000
constexpr size_t OFF_AD    = OFF_AS + SZ_AH;
constexpr size_t OFF_AMAX  = OFF_AD + SZ_AH;                      // encoded uint max
constexpr size_t OFF_DEN   = OFF_AMAX + SZ_AH;
constexpr size_t OFF_ALPHA = OFF_DEN + SZ_AH;                     // exp(alpha) f32 [B][EP][H]
constexpr size_t SZ_ALPHA  = (size_t)BB * EP * HH * 4;            // 10,880,000
constexpr size_t OFF_SPAT  = OFF_ALPHA + SZ_ALPHA;                // aggregated spatial f32 [40000][256]
constexpr size_t SZ_SPAT   = (size_t)ROWS * HID * 4;              // 40,960,000
// total ~106.7 MB

// ---------------- WMMA types / helpers ----------------
typedef __attribute__((ext_vector_type(16))) __bf16 v16bf;
typedef __attribute__((ext_vector_type(8)))  float  v8f;

union FragBF { unsigned int u[8]; v16bf v; };

__device__ __forceinline__ unsigned short f2bf(float f) {
    unsigned u = __float_as_uint(f);
    unsigned r = u + 0x7FFFu + ((u >> 16) & 1u);   // round-to-nearest-even
    return (unsigned short)(r >> 16);
}

__device__ __forceinline__ unsigned fenc(float f) {      // monotone float->uint
    unsigned u = __float_as_uint(f);
    return (u >> 31) ? ~u : (u | 0x80000000u);
}
__device__ __forceinline__ float fdec(unsigned e) {
    unsigned u = (e >> 31) ? (e ^ 0x80000000u) : ~e;
    return __uint_as_float(u);
}
constexpr unsigned ENC_NEG_INF = 0x007FFFFFu;            // fenc(-inf)

__device__ __forceinline__ float sigm(float x) { return 1.f / (1.f + __expf(-x)); }
__device__ __forceinline__ float lrelu(float x) { return x > 0.f ? x : NEG_SLOPE * x; }

// A fragment: 16x32 bf16, row m = lane&15, K-pair dwords split per ISA 7.12.2:
//   lanes 0-15: dw0..3 = K[0..7], dw4..7 = K[16..23]; lanes 16-31: +8
__device__ __forceinline__ v16bf load_a_frag(const unsigned short* __restrict__ Arow,
                                             int k0, int half) {
    FragBF f;
    uint4 a0 = *(const uint4*)(Arow + k0 + 8 * half);
    uint4 a1 = *(const uint4*)(Arow + k0 + 16 + 8 * half);
    f.u[0] = a0.x; f.u[1] = a0.y; f.u[2] = a0.z; f.u[3] = a0.w;
    f.u[4] = a1.x; f.u[5] = a1.y; f.u[6] = a1.z; f.u[7] = a1.w;
    return f.v;
}

// B fragment: 32x16 bf16, col n = lane&15, same K split as A:
// lanes 0-15: K[0..7], K[16..23]; lanes 16-31: +8 (two B128 loads).
__device__ __forceinline__ v16bf load_b_frag(const unsigned short* __restrict__ Brow,
                                             int k0, int half) {
    FragBF f;
    uint4 b0 = *(const uint4*)(Brow + k0 + 8 * half);
    uint4 b1 = *(const uint4*)(Brow + k0 + 16 + 8 * half);
    f.u[0] = b0.x; f.u[1] = b0.y; f.u[2] = b0.z; f.u[3] = b0.w;
    f.u[4] = b1.x; f.u[5] = b1.y; f.u[6] = b1.z; f.u[7] = b1.w;
    return f.v;
}

#define WMMA_BF16(A, Bv, Cacc) \
    __builtin_amdgcn_wmma_f32_16x16x32_bf16(false, (A), false, (Bv), (short)0, (Cacc), false, false)

// ---------------- prep kernels ----------------
__global__ void k_f2bf(const float* __restrict__ in, unsigned short* __restrict__ out, int n) {
    int t = blockIdx.x * blockDim.x + threadIdx.x;
    if (t < n) out[t] = f2bf(in[t]);
}

__global__ void k_build_wcat(const float* __restrict__ W_ih, const float* __restrict__ W_hh,
                             unsigned short* __restrict__ Wcat) {
    int t = blockIdx.x * blockDim.x + threadIdx.x;          // over 1024*512
    if (t >= 4 * HID * 2 * HID) return;
    int n = t >> 9;           // row 0..1023
    int k = t & 511;          // col 0..511
    float v = (k < HID) ? W_ih[(size_t)n * HID + k] : W_hh[(size_t)n * HID + (k - HID)];
    Wcat[t] = f2bf(v);
}

__global__ void k_init(float* __restrict__ spatial, unsigned* __restrict__ amax,
                       float* __restrict__ denom) {
    int t = blockIdx.x * blockDim.x + threadIdx.x;
    if (t < ROWS * HID) spatial[t] = 0.f;
    if (t < ROWS * HH) { amax[t] = ENC_NEG_INF; denom[t] = 0.f; }
}

// ---------------- GEMM 1: xt = x @ lin_w.T  (bf16 WMMA, 32 rows per wave) ----------------
__global__ void k_gemm_lin(const unsigned short* __restrict__ Abf,
                           const unsigned short* __restrict__ Wbf,
                           float* __restrict__ Cout) {
    int wave = blockIdx.x * (blockDim.x >> 5) + (threadIdx.x >> 5);
    if (wave >= (ROWS / 32) * (HID / 16)) return;
    int lane = threadIdx.x & 31;
    int half = lane >> 4, lidx = lane & 15;
    int n0 = (wave & 15) * 16;
    int m0 = (wave >> 4) * 32;

    const unsigned short* Arow0 = Abf + (size_t)(m0 + lidx) * F_IN;
    const unsigned short* Arow1 = Abf + (size_t)(m0 + 16 + lidx) * F_IN;
    const unsigned short* Brow  = Wbf + (size_t)(n0 + lidx) * F_IN;

    v8f acc0 = {}, acc1 = {};
#pragma unroll
    for (int k0 = 0; k0 < F_IN; k0 += 32) {
        v16bf fb  = load_b_frag(Brow, k0, half);
        v16bf fa0 = load_a_frag(Arow0, k0, half);
        acc0 = WMMA_BF16(fa0, fb, acc0);
        v16bf fa1 = load_a_frag(Arow1, k0, half);
        acc1 = WMMA_BF16(fa1, fb, acc1);
    }
#pragma unroll
    for (int r = 0; r < 8; ++r) {
        int row = m0 + r + 8 * half;
        Cout[(size_t)row * HID + n0 + lidx] = acc0[r];
        Cout[(size_t)(row + 16) * HID + n0 + lidx] = acc1[r];
    }
}

// ---------------- attention scores a_s/a_d ----------------
__global__ void k_attn(const float* __restrict__ xt, const float* __restrict__ att_src,
                       const float* __restrict__ att_dst,
                       float* __restrict__ a_s, float* __restrict__ a_d) {
    int t = blockIdx.x * blockDim.x + threadIdx.x;          // over ROWS*H
    if (t >= ROWS * HH) return;
    int row = t >> 2, h = t & 3;
    const float* xrow = xt + (size_t)row * HID + h * CC;
    const float* as = att_src + h * CC;
    const float* ad = att_dst + h * CC;
    float s0 = 0.f, s1 = 0.f;
#pragma unroll 8
    for (int c = 0; c < CC; ++c) { float v = xrow[c]; s0 += v * as[c]; s1 += v * ad[c]; }
    a_s[t] = s0; a_d[t] = s1;
}

// ---------------- segment softmax over edges ----------------
__global__ void k_edge_max(const long long* __restrict__ ei,
                           const float* __restrict__ a_s, const float* __restrict__ a_d,
                           unsigned* __restrict__ amax) {
    int t = blockIdx.x * blockDim.x + threadIdx.x;          // over B*EP
    if (t >= BB * EP) return;
    int b = t / EP, e = t - b * EP;
    int s, d;
    if (e < EE) { s = (int)ei[e]; d = (int)ei[EE + e]; } else { s = d = e - EE; }
    size_t sb = ((size_t)b * NN + s) * HH;
    size_t db = ((size_t)b * NN + d) * HH;
#pragma unroll
    for (int h = 0; h < HH; ++h) {
        float a = lrelu(a_s[sb + h] + a_d[db + h]);
        atomicMax((unsigned*)(amax + db + h), fenc(a));
    }
}

__global__ void k_edge_exp(const long long* __restrict__ ei,
                           const float* __restrict__ a_s, const float* __restrict__ a_d,
                           const unsigned* __restrict__ amax,
                           float* __restrict__ denom, float* __restrict__ alpha) {
    int t = blockIdx.x * blockDim.x + threadIdx.x;          // over B*EP
    if (t >= BB * EP) return;
    int b = t / EP, e = t - b * EP;
    int s, d;
    if (e < EE) { s = (int)ei[e]; d = (int)ei[EE + e]; } else { s = d = e - EE; }
    size_t sb = ((size_t)b * NN + s) * HH;
    size_t db = ((size_t)b * NN + d) * HH;
    size_t ab = ((size_t)b * EP + e) * HH;
#pragma unroll
    for (int h = 0; h < HH; ++h) {
        float a = lrelu(a_s[sb + h] + a_d[db + h]);
        float p = __expf(a - fdec(amax[db + h]));
        alpha[ab + h] = p;
        atomicAdd(denom + db + h, p);
    }
}

// ---------------- message scatter: spatial[dst] += xt[src] * alpha ----------------
__global__ void k_edge_scatter(const long long* __restrict__ ei,
                               const float* __restrict__ alpha, const float* __restrict__ denom,
                               const float* __restrict__ xt, float* __restrict__ spatial) {
    long long t = (long long)blockIdx.x * blockDim.x + threadIdx.x;  // B*EP*H*16 chunks of 4 floats
    if (t >= (long long)BB * EP * HH * 16) return;
    int c4 = (int)(t & 15);
    int h  = (int)((t >> 4) & 3);
    long long te = t >> 6;
    int e = (int)(te % EP);
    int b = (int)(te / EP);
    int s, d;
    if (e < EE) { s = (int)ei[e]; d = (int)ei[EE + e]; } else { s = d = e - EE; }
    float w = alpha[((size_t)b * EP + e) * HH + h] /
              denom[((size_t)b * NN + d) * HH + h];
    const float4 xv = *(const float4*)(xt + ((size_t)b * NN + s) * HID + h * CC + c4 * 4);
    float* dp = spatial + ((size_t)b * NN + d) * HID + h * CC + c4 * 4;
    atomicAdd(dp + 0, xv.x * w);
    atomicAdd(dp + 1, xv.y * w);
    atomicAdd(dp + 2, xv.z * w);
    atomicAdd(dp + 3, xv.w * w);
}

// ---------------- build A2 = bf16([spatial+gat_bias | h_prev]) ----------------
__global__ void k_build_a2(const float* __restrict__ spatial, const float* __restrict__ gat_bias,
                           const float* __restrict__ h_prev, unsigned short* __restrict__ A2) {
    long long t = (long long)blockIdx.x * blockDim.x + threadIdx.x;  // ROWS*512
    if (t >= (long long)ROWS * 512) return;
    int row = (int)(t >> 9);
    int k = (int)(t & 511);
    float v = (k < HID) ? (spatial[(size_t)row * HID + k] + gat_bias[k])
                        : h_prev[(size_t)row * HID + (k - HID)];
    A2[t] = f2bf(v);
}

// ---------------- GEMM 2 + fused LSTM cell (32 rows x 4 gate tiles per wave) ----------------
// gates = A2 @ Wcat.T + b; 8 WMMAs per K-step, B fragments reused across both M tiles.
__global__ void k_gemm_gates(const unsigned short* __restrict__ A2,
                             const unsigned short* __restrict__ Wcat,
                             const float* __restrict__ b_ih, const float* __restrict__ b_hh,
                             const float* __restrict__ c_prev, float* __restrict__ out) {
    int wave = blockIdx.x * (blockDim.x >> 5) + (threadIdx.x >> 5);
    if (wave >= (ROWS / 32) * (HID / 16)) return;
    int lane = threadIdx.x & 31;
    int half = lane >> 4, lidx = lane & 15;
    int n0 = (wave & 15) * 16;
    int m0 = (wave >> 4) * 32;

    const unsigned short* Arow0 = A2 + (size_t)(m0 + lidx) * 512;
    const unsigned short* Arow1 = A2 + (size_t)(m0 + 16 + lidx) * 512;
    const unsigned short* B0 = Wcat + (size_t)(0 * HID + n0 + lidx) * 512;
    const unsigned short* B1 = Wcat + (size_t)(1 * HID + n0 + lidx) * 512;
    const unsigned short* B2 = Wcat + (size_t)(2 * HID + n0 + lidx) * 512;
    const unsigned short* B3 = Wcat + (size_t)(3 * HID + n0 + lidx) * 512;

    v8f ai0 = {}, af0 = {}, ag0 = {}, ao0 = {};   // M tile 0
    v8f ai1 = {}, af1 = {}, ag1 = {}, ao1 = {};   // M tile 1
#pragma unroll 4
    for (int k0 = 0; k0 < 512; k0 += 32) {
        if (k0 + 64 < 512) {
            __builtin_prefetch(Arow0 + k0 + 64, 0, 3);
            __builtin_prefetch(Arow1 + k0 + 64, 0, 3);
        }
        v16bf fa0 = load_a_frag(Arow0, k0, half);
        v16bf fa1 = load_a_frag(Arow1, k0, half);
        v16bf f0 = load_b_frag(B0, k0, half);
        ai0 = WMMA_BF16(fa0, f0, ai0);
        ai1 = WMMA_BF16(fa1, f0, ai1);
        v16bf f1 = load_b_frag(B1, k0, half);
        af0 = WMMA_BF16(fa0, f1, af0);
        af1 = WMMA_BF16(fa1, f1, af1);
        v16bf f2 = load_b_frag(B2, k0, half);
        ag0 = WMMA_BF16(fa0, f2, ag0);
        ag1 = WMMA_BF16(fa1, f2, ag1);
        v16bf f3 = load_b_frag(B3, k0, half);
        ao0 = WMMA_BF16(fa0, f3, ao0);
        ao1 = WMMA_BF16(fa1, f3, ao1);
    }

    int col = n0 + lidx;
    float bi0 = b_ih[0 * HID + col] + b_hh[0 * HID + col];
    float bi1 = b_ih[1 * HID + col] + b_hh[1 * HID + col];
    float bi2 = b_ih[2 * HID + col] + b_hh[2 * HID + col];
    float bi3 = b_ih[3 * HID + col] + b_hh[3 * HID + col];

#pragma unroll
    for (int r = 0; r < 8; ++r) {
        // M tile 0
        {
            int row = m0 + r + 8 * half;
            float iv = sigm(ai0[r] + bi0);
            float fv = sigm(af0[r] + bi1);
            float gv = tanhf(ag0[r] + bi2);
            float ov = sigm(ao0[r] + bi3);
            float cp = c_prev[(size_t)row * HID + col];
            float cn = fv * cp + iv * gv;
            float hn = ov * tanhf(cn);
            out[(size_t)row * HID + col] = hn;                 // slot 0: h_new (pre-LN)
            out[2 * TOT + (size_t)row * HID + col] = cn;       // slot 2: c_new (final)
        }
        // M tile 1
        {
            int row = m0 + 16 + r + 8 * half;
            float iv = sigm(ai1[r] + bi0);
            float fv = sigm(af1[r] + bi1);
            float gv = tanhf(ag1[r] + bi2);
            float ov = sigm(ao1[r] + bi3);
            float cp = c_prev[(size_t)row * HID + col];
            float cn = fv * cp + iv * gv;
            float hn = ov * tanhf(cn);
            out[(size_t)row * HID + col] = hn;
            out[2 * TOT + (size_t)row * HID + col] = cn;
        }
    }
}

// ---------------- LayerNorm (wave per row, wave32 shuffle reduce) ----------------
__global__ void k_layernorm(float* __restrict__ out, const float* __restrict__ ln_g,
                            const float* __restrict__ ln_b) {
    int row = blockIdx.x * (blockDim.x >> 5) + (threadIdx.x >> 5);
    if (row >= ROWS) return;
    int lane = threadIdx.x & 31;
    float* hrow = out + (size_t)row * HID;
    float v[8];
    float4 h0 = *(const float4*)(hrow + lane * 8);
    float4 h1 = *(const float4*)(hrow + lane * 8 + 4);
    v[0] = h0.x; v[1] = h0.y; v[2] = h0.z; v[3] = h0.w;
    v[4] = h1.x; v[5] = h1.y; v[6] = h1.z; v[7] = h1.w;
    float s = 0.f, q = 0.f;
#pragma unroll
    for (int j = 0; j < 8; ++j) { s += v[j]; q += v[j] * v[j]; }
#pragma unroll
    for (int off = 16; off > 0; off >>= 1) {
        s += __shfl_xor(s, off, 32);
        q += __shfl_xor(q, off, 32);
    }
    float mu = s * (1.f / HID);
    float var = q * (1.f / HID) - mu * mu;
    float rinv = rsqrtf(var + LN_EPS);
#pragma unroll
    for (int j = 0; j < 8; ++j) {
        int col = lane * 8 + j;
        float y = (v[j] - mu) * rinv * ln_g[col] + ln_b[col];
        hrow[col] = y;                                     // slot 0: h_ln
        out[TOT + (size_t)row * HID + col] = y;            // slot 1: h_ln
    }
}

// ---------------- host launcher ----------------
extern "C" void kernel_launch(void* const* d_in, const int* in_sizes, int n_in,
                              void* d_out, int out_size, void* d_ws, size_t ws_size,
                              hipStream_t stream) {
    const float*     x        = (const float*)d_in[0];
    const long long* ei       = (const long long*)d_in[1];
    const float*     h_prev   = (const float*)d_in[2];
    const float*     c_prev   = (const float*)d_in[3];
    const float*     lin_w    = (const float*)d_in[4];
    const float*     att_src  = (const float*)d_in[5];
    const float*     att_dst  = (const float*)d_in[6];
    const float*     gat_bias = (const float*)d_in[7];
    const float*     W_ih     = (const float*)d_in[8];
    const float*     W_hh     = (const float*)d_in[9];
    const float*     b_ih     = (const float*)d_in[10];
    const float*     b_hh     = (const float*)d_in[11];
    const float*     ln_g     = (const float*)d_in[12];
    const float*     ln_b     = (const float*)d_in[13];
    float* out = (float*)d_out;

    char* ws = (char*)d_ws;
    unsigned short* xbf     = (unsigned short*)(ws + OFF_XBF);
    unsigned short* wlin    = (unsigned short*)(ws + OFF_WLIN);
    unsigned short* wcat    = (unsigned short*)(ws + OFF_WCAT);
    float*          xt      = (float*)(ws + OFF_XT);
    unsigned short* a2      = (unsigned short*)(ws + OFF_XT);   // aliases xt (dead by then)
    float*          a_s     = (float*)(ws + OFF_AS);
    float*          a_d     = (float*)(ws + OFF_AD);
    unsigned*       amax    = (unsigned*)(ws + OFF_AMAX);
    float*          denom   = (float*)(ws + OFF_DEN);
    float*          alpha   = (float*)(ws + OFF_ALPHA);
    float*          spatial = (float*)(ws + OFF_SPAT);

    const int T = 256;
    // prep: bf16 conversions
    k_f2bf<<<(ROWS * F_IN + T - 1) / T, T, 0, stream>>>(x, xbf, ROWS * F_IN);
    k_f2bf<<<(HID * F_IN + T - 1) / T, T, 0, stream>>>(lin_w, wlin, HID * F_IN);
    k_build_wcat<<<(4 * HID * 2 * HID + T - 1) / T, T, 0, stream>>>(W_ih, W_hh, wcat);
    k_init<<<(ROWS * HID + T - 1) / T, T, 0, stream>>>(spatial, amax, denom);

    // GAT projection (WMMA): 32 rows x 16 cols per wave, 4 waves per block
    k_gemm_lin<<<(ROWS / 32) * (HID / 16) / 4, 128, 0, stream>>>(xbf, wlin, xt);
    k_attn<<<(ROWS * HH + T - 1) / T, T, 0, stream>>>(xt, att_src, att_dst, a_s, a_d);

    // segment softmax + aggregation
    int ethreads = BB * EP;
    k_edge_max<<<(ethreads + T - 1) / T, T, 0, stream>>>(ei, a_s, a_d, amax);
    k_edge_exp<<<(ethreads + T - 1) / T, T, 0, stream>>>(ei, a_s, a_d, amax, denom, alpha);
    long long sthreads = (long long)BB * EP * HH * 16;
    k_edge_scatter<<<(unsigned)((sthreads + T - 1) / T), T, 0, stream>>>(ei, alpha, denom, xt, spatial);

    // LSTM gates (WMMA, fused cell update)
    long long a2threads = (long long)ROWS * 512;
    k_build_a2<<<(unsigned)((a2threads + T - 1) / T), T, 0, stream>>>(spatial, gat_bias, h_prev, a2);
    k_gemm_gates<<<(ROWS / 32) * (HID / 16) / 4, 128, 0, stream>>>(a2, wcat, b_ih, b_hh, c_prev, out);

    // LayerNorm: 8 waves per block, wave per row
    k_layernorm<<<(ROWS + 7) / 8, 256, 0, stream>>>(out, ln_g, ln_b);

    (void)in_sizes; (void)n_in; (void)out_size; (void)ws_size;
}
